// MessagePassing_61383672594925
// MI455X (gfx1250) — matrix operand order repaired and checked by reading, not compiled, not run on hardware
//
#include <hip/hip_runtime.h>
#include <hip/hip_bf16.h>
#include <math.h>

#define N_NODES 30000
#define N_EDGES 480000
#define MUL 32
#define ATTR 8
#define RAD 8
#define FC_H 64

// f16 weight block offsets (in halfs); each matrix stored in FRAGMENT layout:
// fragment (kt,nt) = 32 lanes x 16 halfs contiguous (512 halfs), matching the
// ISA B-operand layout: lane L holds K = (L>>4)*8 + (j&7) + (j>>3)*16 (j=0..15), N = (L&15).
#define OFF_LIN1S 0            // 32x32   (kt=1, nt=2)
#define OFF_LIN1V 1024         // 32x32
#define OFF_SCS   2048         // 256x64  (kt=8, nt=4)
#define OFF_SCV   18432        // 256x32  (kt=8, nt=2)
#define OFF_W1P   26624        // 32x64   (w1 padded K 8->32)
#define OFF_W2    28672        // 64x64   (kt=2, nt=4)
#define OFF_W3    32768        // 64x128  (kt=2, nt=8)
#define OFF_LIN2S 40960        // 64x64
#define OFF_LIN2V 45056        // 64x32
#define WLAYER    47104

typedef __attribute__((ext_vector_type(16))) _Float16 v16h;
typedef __attribute__((ext_vector_type(8)))  float    v8f;

__device__ __forceinline__ float ssp(float x) {
  float ax = fabsf(x);
  float sp = fmaxf(x, 0.0f) + __logf(1.0f + __expf(-ax));
  return sp - 0.6931471805599453f;
}

__device__ __forceinline__ v8f wmma_step(v16h a, v16h b, v8f c) {
  return __builtin_amdgcn_wmma_f32_16x16x32_f16(false, a, false, b, (short)0, c,
                                                false, false);
}

// B fragment: one 32-byte contiguous load per lane from pre-swizzled weights.
__device__ __forceinline__ v16h load_b_frag(const _Float16* seg, int fragidx) {
  int lane = threadIdx.x & 31;
  return *(const v16h*)(seg + (((size_t)fragidx << 5) + lane) * 16);
}

// A fragment from LDS stored in fragment layout (2 x ds_load_b128).
__device__ __forceinline__ v16h load_a_lds(const _Float16* T, int kt) {
  int lane = threadIdx.x & 31;
  return *(const v16h*)(T + (((size_t)kt << 5) + lane) * 16);
}

// Position of element (m, k) of a 16x32 A-tile inside its 512-half fragment.
// Lane L holds M=L&15, K={kb..kb+7, 16+kb..16+kb+7}, kb=8*(L>>4).
__device__ __forceinline__ int a_pos(int m, int k) {
  int h = (k >> 3) & 1;
  int j = (k & 7) | ((k >> 4) << 3);
  return ((m + (h << 4)) << 4) + j;
}

// A fragment built in registers for outer-product tiles (x (x) attr):
// this lane's 16 halves only need two x values and the 8 attr values.
__device__ __forceinline__ v16h a_frag_outer(float x0, float x1, const float* at) {
  v16h a;
#pragma unroll
  for (int j = 0; j < 8; ++j) {
    a[j]     = (_Float16)(x0 * at[j]);
    a[8 + j] = (_Float16)(x1 * at[j]);
  }
  return a;
}

// A fragment from a contiguous 32-float row (this lane's K subset).
__device__ __forceinline__ v16h a_frag_row(const float* row, float scale) {
  int kb = ((threadIdx.x & 31) >> 4) << 3;
  v16h a;
#pragma unroll
  for (int j = 0; j < 8; ++j) {
    a[j]     = (_Float16)(row[kb + j] * scale);
    a[8 + j] = (_Float16)(row[16 + kb + j] * scale);
  }
  return a;
}

// Same but source elements strided by 3 (vector-channel slices).
__device__ __forceinline__ v16h a_frag_row_s3(const float* row, float scale) {
  int kb = ((threadIdx.x & 31) >> 4) << 3;
  v16h a;
#pragma unroll
  for (int j = 0; j < 8; ++j) {
    a[j]     = (_Float16)(row[(kb + j) * 3] * scale);
    a[8 + j] = (_Float16)(row[(16 + kb + j) * 3] * scale);
  }
  return a;
}

// ---------------- weight conversion: f32 -> f16 fragments, norms baked -------
__device__ __forceinline__ _Float16 conv_frag(const float* __restrict__ src,
                                              int Nn, int ntiles, int kvalid,
                                              float scale, int local) {
  int frag = local >> 9, within = local & 511;
  int ln = within >> 4, j = within & 15;
  int kt = frag / ntiles, nt = frag - kt * ntiles;
  int k = kt * 32 + ((ln >> 4) << 3) + (j & 7) + ((j >> 3) << 4);
  int n = nt * 16 + (ln & 15);
  float v = (k < kvalid) ? src[(size_t)k * Nn + n] * scale : 0.0f;
  return (_Float16)v;
}

__global__ void convert_weights_kernel(const float* __restrict__ lin1_s,
                                       const float* __restrict__ lin1_v,
                                       const float* __restrict__ sc_s,
                                       const float* __restrict__ sc_v,
                                       const float* __restrict__ w1,
                                       const float* __restrict__ w2,
                                       const float* __restrict__ w3,
                                       const float* __restrict__ lin2_s,
                                       const float* __restrict__ lin2_v,
                                       _Float16* __restrict__ out) {
  int i = blockIdx.x * blockDim.x + threadIdx.x;
  if (i >= WLAYER) return;
  const float S32 = 0.17677669529663687f;  // 1/sqrt(32)
  const float S8  = 0.35355339059327373f;  // 1/sqrt(8)
  _Float16 r;
  if (i < OFF_LIN1V)      r = conv_frag(lin1_s, 32, 2, 32, S32, i - OFF_LIN1S);
  else if (i < OFF_SCS)   r = conv_frag(lin1_v, 32, 2, 32, S32, i - OFF_LIN1V);
  else if (i < OFF_SCV)   r = conv_frag(sc_s, 64, 4, 256, 0.0625f, i - OFF_SCS);
  else if (i < OFF_W1P)   r = conv_frag(sc_v, 32, 2, 256, 0.0625f, i - OFF_SCV);
  else if (i < OFF_W2)    r = conv_frag(w1, 64, 4, RAD, S8, i - OFF_W1P);
  else if (i < OFF_W3)    r = conv_frag(w2, 64, 4, 64, 0.125f, i - OFF_W2);
  else if (i < OFF_LIN2S) r = conv_frag(w3, 128, 8, 64, 0.125f, i - OFF_W3);
  else if (i < OFF_LIN2V) r = conv_frag(lin2_s, 64, 4, 64, 0.125f, i - OFF_LIN2S);
  else                    r = conv_frag(lin2_v, 32, 2, 64, 0.125f, i - OFF_LIN2V);
  out[i] = r;
}

// ---------------- node prep: s1, v1, sc_sa, sc_va (zero LDS) ------------------
__global__ void __launch_bounds__(128)
node_prep_kernel(const float* __restrict__ x_in,
                 const float* __restrict__ node_attr,
                 const _Float16* __restrict__ wf,
                 float* __restrict__ s1v1,   // (N,128): [0:32]=s1, 32+u*3+c = v1
                 float* __restrict__ scb) {  // (N,160): [0:64]=sc_sa, 64+w*3+c
  int wave = threadIdx.x >> 5, lane = threadIdx.x & 31;
  int tile = blockIdx.x * 4 + wave;
  int base = tile * 16;
  if (base >= N_NODES) return;
  int half = lane >> 4, m = lane & 15, ncol = lane & 15;
  int node = base + m;
  const float* xrow = x_in + (size_t)node * 128;
  float at[8];
#pragma unroll
  for (int a = 0; a < 8; ++a) at[a] = node_attr[(size_t)node * ATTR + a];

  // sc_sa: (s (x) attr) @ sc_s   [16x256]@[256x64]
  float xs0[8], xs1[8];
#pragma unroll
  for (int kt = 0; kt < 8; ++kt) {
    xs0[kt] = xrow[4 * kt + half];
    xs1[kt] = xrow[4 * kt + 2 + half];
  }
  for (int t = 0; t < 4; ++t) {
    v8f c = {};
#pragma unroll
    for (int kt = 0; kt < 8; ++kt)
      c = wmma_step(a_frag_outer(xs0[kt], xs1[kt], at),
                    load_b_frag(wf + OFF_SCS, kt * 4 + t), c);
#pragma unroll
    for (int r = 0; r < 8; ++r)
      scb[(size_t)(base + r + 8 * half) * 160 + (ncol + t * 16)] = c[r];
  }
  // sc_va: (v_c (x) attr) @ sc_v, 3 channels
  for (int ch = 0; ch < 3; ++ch) {
#pragma unroll
    for (int kt = 0; kt < 8; ++kt) {
      xs0[kt] = xrow[32 + (4 * kt + half) * 3 + ch];
      xs1[kt] = xrow[32 + (4 * kt + 2 + half) * 3 + ch];
    }
    for (int t = 0; t < 2; ++t) {
      v8f c = {};
#pragma unroll
      for (int kt = 0; kt < 8; ++kt)
        c = wmma_step(a_frag_outer(xs0[kt], xs1[kt], at),
                      load_b_frag(wf + OFF_SCV, kt * 2 + t), c);
#pragma unroll
      for (int r = 0; r < 8; ++r)
        scb[(size_t)(base + r + 8 * half) * 160 + 64 + (ncol + t * 16) * 3 + ch] = c[r];
    }
  }
  // s1 = s @ lin1_s
  for (int t = 0; t < 2; ++t) {
    v8f c = {};
    c = wmma_step(a_frag_row(xrow, 1.0f), load_b_frag(wf + OFF_LIN1S, t), c);
#pragma unroll
    for (int r = 0; r < 8; ++r)
      s1v1[(size_t)(base + r + 8 * half) * 128 + (ncol + t * 16)] = c[r];
  }
  // v1 per channel
  for (int ch = 0; ch < 3; ++ch) {
    for (int t = 0; t < 2; ++t) {
      v8f c = {};
      c = wmma_step(a_frag_row_s3(xrow + 32 + ch, 1.0f),
                    load_b_frag(wf + OFF_LIN1V, t), c);
#pragma unroll
      for (int r = 0; r < 8; ++r)
        s1v1[(size_t)(base + r + 8 * half) * 128 + 32 + (ncol + t * 16) * 3 + ch] = c[r];
    }
  }
}

// ---------------- edge kernel: MLP (WMMA) + fused message scatter -------------
__global__ void __launch_bounds__(128)
edge_kernel(const int* __restrict__ edge_src, const int* __restrict__ edge_dst,
            const float* __restrict__ edge_attr,
            const float* __restrict__ edge_scalars,
            const float* __restrict__ s1v1, const _Float16* __restrict__ wf,
            float* __restrict__ agg) {
  __shared__ __align__(32) _Float16 sA[4][1024];  // 16x64 in A-fragment layout
  __shared__ __align__(32) _Float16 sB[4][1024];
  __shared__ float sW[4][16 * 128];
  int wave = threadIdx.x >> 5, lane = threadIdx.x & 31;
  int tile = blockIdx.x * 4 + wave;
  int ebase = tile * 16;
  if (ebase >= N_EDGES) return;
  _Float16* A = sA[wave]; _Float16* B = sB[wave]; float* W = sW[wave];
  int half = lane >> 4, ncol = lane & 15, el = lane & 15;

  // h0 A-fragment in registers (K padded 8->32; only half0 lanes carry data)
  v16h a0;
  {
    const float* es = edge_scalars + (size_t)(ebase + el) * RAD;
#pragma unroll
    for (int j = 0; j < 8; ++j) {
      a0[j]     = (_Float16)(half == 0 ? es[j] : 0.0f);
      a0[8 + j] = (_Float16)0.0f;
    }
  }
  // h0 = ssp(scalars @ w1)  -> B in A-fragment layout
  for (int t = 0; t < 4; ++t) {
    v8f c = {};
    c = wmma_step(a0, load_b_frag(wf + OFF_W1P, t), c);
#pragma unroll
    for (int r = 0; r < 8; ++r) {
      int mm = r + 8 * half, kg = ncol + 16 * t;
      B[(kg >> 5) * 512 + a_pos(mm, kg & 31)] = (_Float16)ssp(c[r]);
    }
  }
  // h1 = ssp(h0 @ w2)  -> A in A-fragment layout
  for (int t = 0; t < 4; ++t) {
    v8f c = {};
#pragma unroll
    for (int ks = 0; ks < 2; ++ks)
      c = wmma_step(load_a_lds(B, ks), load_b_frag(wf + OFF_W2, ks * 4 + t), c);
#pragma unroll
    for (int r = 0; r < 8; ++r) {
      int mm = r + 8 * half, kg = ncol + 16 * t;
      A[(kg >> 5) * 512 + a_pos(mm, kg & 31)] = (_Float16)ssp(c[r]);
    }
  }
  // w = h1 @ w3  -> W (16x128 f32): [p*32+u]
  for (int t = 0; t < 8; ++t) {
    v8f c = {};
#pragma unroll
    for (int ks = 0; ks < 2; ++ks)
      c = wmma_step(load_a_lds(A, ks), load_b_frag(wf + OFF_W3, ks * 8 + t), c);
#pragma unroll
    for (int r = 0; r < 8; ++r)
      W[(r + 8 * half) * 128 + ncol + t * 16] = c[r];
  }

  // fused message formation + scatter (agg is L2-resident)
  int e = ebase + el;
  int src = edge_src[e];
  int dst = edge_dst[e];
  float y0 = edge_attr[(size_t)e * 4 + 0];
  float y1 = edge_attr[(size_t)e * 4 + 1];
  float y2 = edge_attr[(size_t)e * 4 + 2];
  float y3 = edge_attr[(size_t)e * 4 + 3];
  const float* srow = s1v1 + (size_t)src * 128;
  float* arow = agg + (size_t)dst * 256;
#pragma unroll 4
  for (int uu = 0; uu < 16; ++uu) {
    int u = half * 16 + uu;
    float se  = srow[u];
    float ve0 = srow[32 + u * 3 + 0];
    float ve1 = srow[32 + u * 3 + 1];
    float ve2 = srow[32 + u * 3 + 2];
    float w0  = W[el * 128 + u];
    float w1v = W[el * 128 + 32 + u];
    float w2v = W[el * 128 + 64 + u];
    float w3v = W[el * 128 + 96 + u];
    float dot = ve0 * y1 + ve1 * y2 + ve2 * y3;
    atomicAdd(&arow[u], w0 * se * y0);              // m_s1
    atomicAdd(&arow[32 + u], w3v * dot);            // m_s2
    float ws1 = w1v * se;                           // m_v1
    atomicAdd(&arow[64 + u * 3 + 0], ws1 * y1);
    atomicAdd(&arow[64 + u * 3 + 1], ws1 * y2);
    atomicAdd(&arow[64 + u * 3 + 2], ws1 * y3);
    float wy0 = w2v * y0;                           // m_v2
    atomicAdd(&arow[160 + u * 3 + 0], wy0 * ve0);
    atomicAdd(&arow[160 + u * 3 + 1], wy0 * ve1);
    atomicAdd(&arow[160 + u * 3 + 2], wy0 * ve2);
  }
}

// ---------------- node output: lin2 GEMMs + gating + residual ----------------
__global__ void __launch_bounds__(128)
node_out_kernel(const float* __restrict__ x_in, const float* __restrict__ agg,
                const float* __restrict__ scb, const _Float16* __restrict__ wf,
                float* __restrict__ x_out) {
  __shared__ float sOS[4][16 * 64];
  int wave = threadIdx.x >> 5, lane = threadIdx.x & 31;
  int tile = blockIdx.x * 4 + wave;
  int base = tile * 16;
  if (base >= N_NODES) return;
  float* OS = sOS[wave];
  int half = lane >> 4, m = lane & 15, ncol = lane & 15;
  const float* arow = agg + (size_t)(base + m) * 256;

  // out_s = (agg[:, :64]/sqrt(16)) @ lin2_s + sc_sa
  for (int t = 0; t < 4; ++t) {
    v8f c = {};
#pragma unroll
    for (int ks = 0; ks < 2; ++ks)
      c = wmma_step(a_frag_row(arow + ks * 32, 0.25f),
                    load_b_frag(wf + OFF_LIN2S, ks * 4 + t), c);
#pragma unroll
    for (int r = 0; r < 8; ++r) {
      int mm = r + 8 * half, col = ncol + t * 16;
      OS[mm * 64 + col] = c[r] + scb[(size_t)(base + mm) * 160 + col];
    }
  }
  // scalar outputs with residual
  for (int idx = lane; idx < 16 * 32; idx += 32) {
    int mm = idx >> 5, u = idx & 31;
    int node = base + mm;
    x_out[(size_t)node * 128 + u] =
        x_in[(size_t)node * 128 + u] + ssp(OS[mm * 64 + u]);
  }
  // vector outputs: per channel a_v_c @ lin2_v + sc_va, gated, residual
  for (int ch = 0; ch < 3; ++ch) {
    for (int t = 0; t < 2; ++t) {
      v8f c = {};
#pragma unroll
      for (int ks = 0; ks < 2; ++ks)
        c = wmma_step(a_frag_row_s3(arow + 64 + ch + ks * 96, 0.25f),
                      load_b_frag(wf + OFF_LIN2V, ks * 2 + t), c);
#pragma unroll
      for (int r = 0; r < 8; ++r) {
        int mm = r + 8 * half, w = ncol + t * 16;
        int node = base + mm;
        float ov = c[r] + scb[(size_t)node * 160 + 64 + w * 3 + ch];
        float gate = ssp(OS[mm * 64 + 32 + w]);
        size_t o = (size_t)node * 128 + 32 + w * 3 + ch;
        x_out[o] = x_in[o] + ov * gate;
      }
    }
  }
}

extern "C" void kernel_launch(void* const* d_in, const int* in_sizes, int n_in,
                              void* d_out, int out_size, void* d_ws, size_t ws_size,
                              hipStream_t stream) {
  const float* node_input   = (const float*)d_in[0];
  const float* node_attr    = (const float*)d_in[1];
  const int*   edge_src     = (const int*)d_in[2];
  const int*   edge_dst     = (const int*)d_in[3];
  const float* edge_attr    = (const float*)d_in[4];
  const float* edge_scalars = (const float*)d_in[5];
  const float* lin1_s = (const float*)d_in[6];
  const float* lin1_v = (const float*)d_in[7];
  const float* mlp_w1 = (const float*)d_in[8];
  const float* mlp_w2 = (const float*)d_in[9];
  const float* mlp_w3 = (const float*)d_in[10];
  const float* lin2_s = (const float*)d_in[11];
  const float* lin2_v = (const float*)d_in[12];
  const float* sc_s   = (const float*)d_in[13];
  const float* sc_v   = (const float*)d_in[14];

  float* ws   = (float*)d_ws;
  float* xA   = ws;                               // (N,128)
  float* xB   = xA + (size_t)N_NODES * 128;       // (N,128)
  float* s1v1 = xB + (size_t)N_NODES * 128;       // (N,128)
  float* scb  = s1v1 + (size_t)N_NODES * 128;     // (N,160)
  float* agg  = scb + (size_t)N_NODES * 160;      // (N,256)
  _Float16* wf16 = (_Float16*)(agg + (size_t)N_NODES * 256);

  for (int l = 0; l < 3; ++l) {
    convert_weights_kernel<<<(WLAYER + 255) / 256, 256, 0, stream>>>(
        lin1_s + (size_t)l * 32 * 32, lin1_v + (size_t)l * 32 * 32,
        sc_s + (size_t)l * 256 * 64, sc_v + (size_t)l * 256 * 32,
        mlp_w1 + (size_t)l * RAD * FC_H, mlp_w2 + (size_t)l * FC_H * FC_H,
        mlp_w3 + (size_t)l * FC_H * 128, lin2_s + (size_t)l * 64 * 64,
        lin2_v + (size_t)l * 64 * 32, wf16 + (size_t)l * WLAYER);
  }

  const int node_blocks = (N_NODES / 16 + 3) / 4;
  const int edge_blocks = N_EDGES / 64;

  const float* xin = node_input;
  for (int l = 0; l < 3; ++l) {
    float* xout = (l == 0) ? xA : (l == 1) ? xB : (float*)d_out;
    const _Float16* wf = wf16 + (size_t)l * WLAYER;
    hipMemsetAsync(agg, 0, (size_t)N_NODES * 256 * sizeof(float), stream);
    node_prep_kernel<<<node_blocks, 128, 0, stream>>>(xin, node_attr, wf, s1v1, scb);
    edge_kernel<<<edge_blocks, 128, 0, stream>>>(edge_src, edge_dst, edge_attr,
                                                 edge_scalars, s1v1, wf, agg);
    node_out_kernel<<<node_blocks, 128, 0, stream>>>(xin, agg, scb, wf, xout);
    xin = xout;
  }
}
